// TriangleCrossMultiplicativeUpdate_56023553409070
// MI455X (gfx1250) — compile-verified, matrix operand order checked by reading, not run on hardware
//
#include <hip/hip_runtime.h>
#include <hip/hip_bf16.h>

// MI455X / gfx1250 fused TriangleCrossMultiplicativeUpdate.
// All dense contractions use v_wmma_f32_16x16x32_bf16 (fp32 accumulate).
// One workgroup (256 threads = 8 wave32) per node n; wave w owns channel
// tile z in [16w, 16w+16).

typedef __attribute__((ext_vector_type(16))) __bf16 v16bf;
typedef __attribute__((ext_vector_type(8)))  float  v8f;

#define N_NODES 1024
#define KSLOT   16
#define CSF     256
#define CZF     128
#define GF      16
#define NRBF    64

// CDNA5 16-bit A-matrix 16x32 per-lane K pattern (CDNA5 ISA):
// lane group g = lane/16; VGPR v holds K pair (2 elems). v<4: K = 2v + 8g + p
// ; v>=4: K = 16 + 2(v-4) + 8g + p.
__device__ __forceinline__ int kpat(int h, int g) {
  int v = h >> 1, p = h & 1;
  int base = (v < 4) ? (v * 2) : (16 + (v - 4) * 2);
  return base + g * 8 + p;
}

__device__ __forceinline__ float sigmoidf(float x) {
  return 1.0f / (1.0f + __expf(-x));
}

__device__ __forceinline__ v8f zero8() {
  v8f z;
#pragma unroll
  for (int i = 0; i < 8; ++i) z[i] = 0.0f;
  return z;
}

__device__ __forceinline__ v8f wmma_bf16(v16bf a, v16bf b, v8f c) {
  return __builtin_amdgcn_wmma_f32_16x16x32_bf16(
      /*neg_a=*/false, a, /*neg_b=*/false, b,
      /*c_mod=*/(short)0, c, /*reuse_a=*/false, /*reuse_b=*/false);
}

// ---------------------------------------------------------------------------
// Kernel A: node_left / node_right projections [1024,256] -> [1024,16] each.
// ws layout: [0 .. 16383] node_left (n*16+g), [16384 .. 32767] node_right.
// ---------------------------------------------------------------------------
__global__ __launch_bounds__(256) void node_proj_kernel(
    const float* __restrict__ nodeF,
    const float* __restrict__ W_nl, const float* __restrict__ b_nl,
    const float* __restrict__ W_nr, const float* __restrict__ b_nr,
    float* __restrict__ ws) {
  int t = blockIdx.x * 256 + threadIdx.x;       // 0 .. 32767
  int which = t >> 14;                          // 0 = left, 1 = right
  int n  = (t >> 4) & (N_NODES - 1);
  int gg = t & 15;
  const float* W = which ? W_nr : W_nl;
  const float* b = which ? b_nr : b_nl;
  const float4* xr = (const float4*)(nodeF + (size_t)n * CSF);
  const float4* wr = (const float4*)(W + (size_t)gg * CSF);
  float s = 0.0f;
#pragma unroll 4
  for (int c = 0; c < CSF / 4; ++c) {
    float4 a = xr[c], bb = wr[c];
    s += a.x * bb.x + a.y * bb.y + a.z * bb.z + a.w * bb.w;
  }
  ws[t] = s + b[gg];
}

// ---------------------------------------------------------------------------
// Kernel C: fully fused triangle update, one block per node n.
// ---------------------------------------------------------------------------
__global__ __launch_bounds__(256) void triangle_fused_kernel(
    const float* __restrict__ trans,
    const float* __restrict__ srcF,  const float* __restrict__ dstF,
    const float* __restrict__ ln_src_g, const float* __restrict__ ln_src_b,
    const float* __restrict__ ln_dst_g, const float* __restrict__ ln_dst_b,
    const float* __restrict__ W_ep, const float* __restrict__ b_ep,
    const float* __restrict__ W_eg, const float* __restrict__ b_eg,
    const float* __restrict__ W_dg, const float* __restrict__ b_dg,
    const float* __restrict__ W_dp, const float* __restrict__ b_dp,
    const float* __restrict__ ln_out_g, const float* __restrict__ ln_out_b,
    const float* __restrict__ W_lo, const float* __restrict__ b_lo,
    const float* __restrict__ W_og, const float* __restrict__ b_og,
    const int* __restrict__ src_idx0, const int* __restrict__ dst_idx0,
    const unsigned char* __restrict__ src_mask,
    const unsigned char* __restrict__ dst_mask,
    const float* __restrict__ node_lr,       // ws from kernel A
    float* __restrict__ out) {

  __shared__ __bf16 sln[16][136];     // LN(src_edge_features) bf16
  __shared__ __bf16 dln[16][136];     // LN(dst_edge_features) bf16
  __shared__ float  e1s[16][16];      // gathered node_left
  __shared__ float  e2s[16][16];      // gathered node_right
  __shared__ float  t1s[16][4];
  __shared__ float  t2s[16][4];
  __shared__ __bf16 phi[256][66];     // RBF features [ij][r] bf16
  __shared__ float  edge2s[16][132];  // edge2[j][z]
  __shared__ float  ogs[16][132];     // sigmoid og gate [i][z]
  __shared__ float  upds[16][132];    // upd accumulator [i][z]
  __shared__ __bf16 updn[16][136];    // LN(upd) bf16
  __shared__ float  smk[16], dmk[16];

  const int n   = blockIdx.x;
  const int tid = threadIdx.x;
  const int w   = tid >> 5;       // wave / N-tile id (z tile)
  const int l   = tid & 31;       // lane
  const int gl  = l >> 4;         // lane group (0/1)
  const int m   = l & 15;         // lane within group

  // ---- Phase 1: gathers ---------------------------------------------------
  {
    int i  = tid >> 4, gg = tid & 15;
    int si = src_idx0[n * KSLOT + i];
    int di = dst_idx0[n * KSLOT + i];
    e1s[i][gg] = node_lr[si * GF + gg];
    e2s[i][gg] = node_lr[16384 + di * GF + gg];
    if (gg < 3) {
      t1s[i][gg] = trans[si * 3 + gg];
      t2s[i][gg] = trans[di * 3 + gg];
    }
    if (gg == 0) {
      smk[i] = src_mask[n * KSLOT + i] ? 1.0f : 0.0f;
      dmk[i] = dst_mask[n * KSLOT + i] ? 1.0f : 0.0f;
    }
  }

  // ---- Phase 2: LayerNorm of src/dst edge features (row = 16 lanes) ------
  {
    int r = w * 2 + gl;   // row 0..15, one 16-lane half-wave per row
    // src
    float x[8], s1 = 0.0f, s2 = 0.0f;
#pragma unroll
    for (int q = 0; q < 8; ++q) {
      x[q] = srcF[((size_t)n * KSLOT + r) * CZF + m + q * 16];
      s1 += x[q]; s2 += x[q] * x[q];
    }
#pragma unroll
    for (int mk = 8; mk >= 1; mk >>= 1) {
      s1 += __shfl_xor(s1, mk, 32);
      s2 += __shfl_xor(s2, mk, 32);
    }
    float mean = s1 * (1.0f / CZF);
    float rstd = rsqrtf(s2 * (1.0f / CZF) - mean * mean + 1e-5f);
#pragma unroll
    for (int q = 0; q < 8; ++q) {
      int c = m + q * 16;
      sln[r][c] = (__bf16)((x[q] - mean) * rstd * ln_src_g[c] + ln_src_b[c]);
    }
    // dst
    s1 = 0.0f; s2 = 0.0f;
#pragma unroll
    for (int q = 0; q < 8; ++q) {
      x[q] = dstF[((size_t)n * KSLOT + r) * CZF + m + q * 16];
      s1 += x[q]; s2 += x[q] * x[q];
    }
#pragma unroll
    for (int mk = 8; mk >= 1; mk >>= 1) {
      s1 += __shfl_xor(s1, mk, 32);
      s2 += __shfl_xor(s2, mk, 32);
    }
    mean = s1 * (1.0f / CZF);
    rstd = rsqrtf(s2 * (1.0f / CZF) - mean * mean + 1e-5f);
#pragma unroll
    for (int q = 0; q < 8; ++q) {
      int c = m + q * 16;
      dln[r][c] = (__bf16)((x[q] - mean) * rstd * ln_dst_g[c] + ln_dst_b[c]);
    }
  }
  __syncthreads();

  const int z = w * 16 + m;   // this lane's output channel

  // ---- Phase 3: edge2 / og projections ([16,128]@[128,16] per wave) -----
  {
    v8f aeg = zero8(), aep = zero8(), aog = zero8();
#pragma unroll
    for (int ks = 0; ks < 4; ++ks) {
      v16bf As, Ad, Beg, Bep, Bog;
#pragma unroll
      for (int h = 0; h < 16; ++h) {
        int k = ks * 32 + kpat(h, gl);
        As[h]  = sln[m][k];
        Ad[h]  = dln[m][k];
        Beg[h] = (__bf16)W_eg[z * CZF + k];
        Bep[h] = (__bf16)W_ep[z * CZF + k];
        Bog[h] = (__bf16)W_og[z * CZF + k];
      }
      aeg = wmma_bf16(As, Beg, aeg);
      aep = wmma_bf16(As, Bep, aep);
      aog = wmma_bf16(Ad, Bog, aog);
    }
    float beg = b_eg[z], bep = b_ep[z], bog = b_og[z];
#pragma unroll
    for (int r = 0; r < 8; ++r) {
      int row = r + 8 * gl;
      edge2s[row][z] = sigmoidf(aeg[r] + beg) * (aep[r] + bep);
      ogs[row][z]    = sigmoidf(aog[r] + bog);
    }
  }

  // ---- Phase 4: RBF features phi[ij][r] -----------------------------------
  {
    int i = tid >> 4, j = tid & 15;
    float dx = t1s[i][0] - t2s[j][0] + 1e-8f;
    float dy = t1s[i][1] - t2s[j][1] + 1e-8f;
    float dz = t1s[i][2] - t2s[j][2] + 1e-8f;
    float d  = sqrtf(dx * dx + dy * dy + dz * dz);
#pragma unroll 8
    for (int r = 0; r < NRBF; ++r) {
      float mu = 20.0f * (float)r * (1.0f / 63.0f);
      float a  = (d - mu) * 3.2f;               // / sigma, sigma = 20/64
      phi[tid][r] = (__bf16)__expf(-a * a);
    }
  }

  // ---- Phase 5: cache W_dg / W_dp B-fragments in registers ---------------
  v16bf Bg[8], Bd[2];
#pragma unroll
  for (int ks = 0; ks < 8; ++ks)
#pragma unroll
    for (int h = 0; h < 16; ++h)
      Bg[ks][h] = (__bf16)W_dg[z * 256 + ks * 32 + kpat(h, gl)];
#pragma unroll
  for (int ks = 0; ks < 2; ++ks)
#pragma unroll
    for (int h = 0; h < 16; ++h)
      Bd[ks][h] = (__bf16)W_dp[z * NRBF + ks * 32 + kpat(h, gl)];

  __syncthreads();

  // ---- Phase 6: triangle GEMMs + fused j-reduction ------------------------
  // gate3 : M=ij(256), K=ab(256), N=z(128), A built on the fly from e1*e2.
  // distf : M=ij(256), K=r(64),  N=z(128), A from phi LDS.
  {
    float bdg = b_dg[z], bdp = b_dp[z];
    for (int mt = 0; mt < 16; ++mt) {   // M tile <=> src slot i = mt
      v8f ag = zero8(), ad = zero8();
#pragma unroll
      for (int ks = 0; ks < 8; ++ks) {
        v16bf A;
#pragma unroll
        for (int h = 0; h < 16; ++h) {
          int k = ks * 32 + kpat(h, gl);          // k = a*16 + b
          A[h] = (__bf16)(e1s[mt][k >> 4] * e2s[m][k & 15]);
        }
        ag = wmma_bf16(A, Bg[ks], ag);
      }
#pragma unroll
      for (int ks = 0; ks < 2; ++ks) {
        v16bf A;
#pragma unroll
        for (int h = 0; h < 16; ++h)
          A[h] = phi[mt * 16 + m][ks * 32 + kpat(h, gl)];
        ad = wmma_bf16(A, Bd[ks], ad);
      }
      // edge3 = sigmoid(gate3+b_dg) * (distf+b_dp) * mask3 ; reduce over j
      float part = 0.0f;
      float sm = smk[mt];
#pragma unroll
      for (int r = 0; r < 8; ++r) {
        int j = r + 8 * gl;                       // D-frag row = dst slot j
        float e3 = sigmoidf(ag[r] + bdg) * (ad[r] + bdp) * sm * dmk[j];
        part += e3 * edge2s[j][z];
      }
      part += __shfl_xor(part, 16, 32);           // join j halves (wave32)
      if (gl == 0) upds[mt][z] = part;
    }
  }
  __syncthreads();

  // ---- Phase 7: LayerNorm(upd) -> bf16 ------------------------------------
  {
    int r = w * 2 + gl;
    float x[8], s1 = 0.0f, s2 = 0.0f;
#pragma unroll
    for (int q = 0; q < 8; ++q) {
      x[q] = upds[r][m + q * 16];
      s1 += x[q]; s2 += x[q] * x[q];
    }
#pragma unroll
    for (int mk = 8; mk >= 1; mk >>= 1) {
      s1 += __shfl_xor(s1, mk, 32);
      s2 += __shfl_xor(s2, mk, 32);
    }
    float mean = s1 * (1.0f / CZF);
    float rstd = rsqrtf(s2 * (1.0f / CZF) - mean * mean + 1e-5f);
#pragma unroll
    for (int q = 0; q < 8; ++q) {
      int c = m + q * 16;
      updn[r][c] = (__bf16)((x[q] - mean) * rstd * ln_out_g[c] + ln_out_b[c]);
    }
  }
  __syncthreads();

  // ---- Phase 8: final projection W_lo, og gate, dst mask, store ----------
  {
    v8f al = zero8();
#pragma unroll
    for (int ks = 0; ks < 4; ++ks) {
      v16bf A, B;
#pragma unroll
      for (int h = 0; h < 16; ++h) {
        int k = ks * 32 + kpat(h, gl);
        A[h] = updn[m][k];
        B[h] = (__bf16)W_lo[z * CZF + k];
      }
      al = wmma_bf16(A, B, al);
    }
    float bl = b_lo[z];
#pragma unroll
    for (int r = 0; r < 8; ++r) {
      int row = r + 8 * gl;                       // slot i
      float v = (al[r] + bl) * ogs[row][z] * dmk[row];
      out[((size_t)n * KSLOT + row) * CZF + z] = v;
    }
  }
}

// ---------------------------------------------------------------------------
// Launch. Input order follows setup_inputs(). ws needs 128 KB (node_left/right).
// ---------------------------------------------------------------------------
extern "C" void kernel_launch(void* const* d_in, const int* in_sizes, int n_in,
                              void* d_out, int out_size, void* d_ws, size_t ws_size,
                              hipStream_t stream) {
  const float* node_features = (const float*)d_in[0];
  const float* trans         = (const float*)d_in[1];
  const float* srcF          = (const float*)d_in[2];
  const float* dstF          = (const float*)d_in[3];
  const float* ln_src_g      = (const float*)d_in[4];
  const float* ln_src_b      = (const float*)d_in[5];
  const float* ln_dst_g      = (const float*)d_in[6];
  const float* ln_dst_b      = (const float*)d_in[7];
  const float* W_nl          = (const float*)d_in[8];
  const float* b_nl          = (const float*)d_in[9];
  const float* W_nr          = (const float*)d_in[10];
  const float* b_nr          = (const float*)d_in[11];
  const float* W_ep          = (const float*)d_in[12];
  const float* b_ep          = (const float*)d_in[13];
  const float* W_eg          = (const float*)d_in[14];
  const float* b_eg          = (const float*)d_in[15];
  const float* W_dg          = (const float*)d_in[16];
  const float* b_dg          = (const float*)d_in[17];
  const float* W_dp          = (const float*)d_in[18];
  const float* b_dp          = (const float*)d_in[19];
  const float* ln_out_g      = (const float*)d_in[20];
  const float* ln_out_b      = (const float*)d_in[21];
  const float* W_lo          = (const float*)d_in[22];
  const float* b_lo          = (const float*)d_in[23];
  const float* W_og          = (const float*)d_in[24];
  const float* b_og          = (const float*)d_in[25];
  const int*   src_idx       = (const int*)d_in[26];   // [2,N,K]; use row 0
  const int*   dst_idx       = (const int*)d_in[27];
  const unsigned char* src_mask = (const unsigned char*)d_in[28];
  const unsigned char* dst_mask = (const unsigned char*)d_in[29];

  float* ws  = (float*)d_ws;        // 2 * 1024 * 16 floats = 128 KB
  float* out = (float*)d_out;

  node_proj_kernel<<<(2 * N_NODES * GF) / 256, 256, 0, stream>>>(
      node_features, W_nl, b_nl, W_nr, b_nr, ws);

  triangle_fused_kernel<<<N_NODES, 256, 0, stream>>>(
      trans, srcF, dstF,
      ln_src_g, ln_src_b, ln_dst_g, ln_dst_b,
      W_ep, b_ep, W_eg, b_eg, W_dg, b_dg, W_dp, b_dp,
      ln_out_g, ln_out_b, W_lo, b_lo, W_og, b_og,
      src_idx, dst_idx, src_mask, dst_mask,
      ws, out);
}
